// SeedOssAttention_36318243455436
// MI455X (gfx1250) — matrix operand order repaired and checked by reading, not compiled, not run on hardware
//
#include <hip/hip_runtime.h>

// ---------------------------------------------------------------------------
// SeedOss GQA attention block for MI455X (gfx1250).
// bf16 WMMA, TDM (tensor_load_to_lds) double-buffered tile staging with
// async-to-LDS fallback, exp2-domain online softmax.
// Shapes: B=2,S=2048,H=4096,NH=32,NKV=8,D=128.
// ---------------------------------------------------------------------------

typedef __bf16 bf16_t;
typedef __attribute__((ext_vector_type(16))) __bf16 v16bf;
typedef __attribute__((ext_vector_type(8)))  float  v8f;
typedef __attribute__((ext_vector_type(4)))  unsigned int v4u;
typedef __attribute__((ext_vector_type(8)))  int v8i;
typedef __attribute__((ext_vector_type(4)))  int v4i;

namespace cfg {
constexpr int B   = 2;
constexpr int S   = 2048;
constexpr int H   = 4096;
constexpr int NH  = 32;
constexpr int NKV = 8;
constexpr int D   = 128;
constexpr int GROUPS = NH / NKV;
constexpr int TOK = B * S;
// 1/sqrt(128) * log2(e): folded into Q; softmax runs in exp2 domain.
constexpr float QSCALE = 0.08838834764831845f * 1.4426950408889634f;
constexpr float NEG = -1e30f;
}

#define USE_TDM __has_builtin(__builtin_amdgcn_tensor_load_to_lds)

__device__ __forceinline__ v8f wmma_bf16(v16bf a, v16bf b, v8f c) {
  return __builtin_amdgcn_wmma_f32_16x16x32_bf16(
      false, a, false, b, (short)0, c, false, false);
}

// gfx1250 async copy fallback: 16B per lane, global -> LDS (ASYNCcnt).
__device__ __forceinline__ void async_ld_b128(void* lds, const void* gaddr) {
  unsigned off = (unsigned)(uintptr_t)lds;
  asm volatile("global_load_async_to_lds_b128 %0, %1, off"
               :: "v"(off), "v"(gaddr) : "memory");
}
__device__ __forceinline__ void wait_async0() {
  asm volatile("s_wait_asynccnt 0x0" ::: "memory");
}

#if USE_TDM
// One-instruction 2D tile DMA: tile_x contiguous elems per row, tile_y rows,
// row pitch stride_elems; bf16 (data_size=1 -> 2 bytes). D# per ISA ch.8.
// This toolchain exposes the 6-arg builtin:
//   (uint32x4 g0, int32x8 g1, int32x4 g2, int32x4 g3, int32x8 g4, i32 cpol)
// Groups 2..4 describe dims >2 / gather indices; zero for a 2-D tile.
__device__ __forceinline__ void tdm_load_2d(
    unsigned lds_addr, const void* gaddr,
    unsigned tile_x, unsigned tile_y, unsigned stride_elems) {
  unsigned long long ga = (unsigned long long)(uintptr_t)gaddr;
  v4u g0;
  g0[0] = 1u;                                    // count=1, user descriptor
  g0[1] = lds_addr;                              // LDS byte address
  g0[2] = (unsigned)(ga & 0xFFFFFFFFu);          // global_addr[31:0]
  g0[3] = (unsigned)((ga >> 32) & 0x1FFFFFFu)    // global_addr[56:32]
          | (2u << 30);                          // type = 2 ("image")
  const unsigned td0 = 1u << 20, td1 = 1u << 20; // generous OOB bounds
  v8i g1;
  g1[0] = (int)(1u << 16);                       // data_size=1 (2B elements)
  g1[1] = (int)((td0 & 0xFFFFu) << 16);          // tensor_dim0[15:0]
  g1[2] = (int)((td0 >> 16) | ((td1 & 0xFFFFu) << 16));
  g1[3] = (int)((td1 >> 16) | (tile_x << 16));   // tile_dim0
  g1[4] = (int)tile_y;                           // tile_dim1 (tile_dim2=0)
  g1[5] = (int)stride_elems;                     // tensor_dim0_stride[31:0]
  g1[6] = 0;
  g1[7] = 0;
  v4i z4 = {0, 0, 0, 0};
  v8i z8 = {0, 0, 0, 0, 0, 0, 0, 0};
  __builtin_amdgcn_tensor_load_to_lds(g0, g1, z4, z4, z8, 0);
}
__device__ __forceinline__ void wait_tensor0() {
  __builtin_amdgcn_s_wait_tensorcnt((short)0);
}
#endif

__device__ __forceinline__ void ld_frag(v16bf& dst, const bf16_t* lo, const bf16_t* hi) {
  ((uint4*)&dst)[0] = *(const uint4*)lo;
  ((uint4*)&dst)[1] = *(const uint4*)hi;
}
__device__ __forceinline__ float fast_exp2(float x) { return __builtin_amdgcn_exp2f(x); }
__device__ __forceinline__ float fast_rcp(float x)  { return __builtin_amdgcn_rcpf(x); }

// ---------------------------------------------------------------------------
// Kernel 1a: plain f32 -> bf16 (activations)
// ---------------------------------------------------------------------------
__global__ __launch_bounds__(256) void cvt_bf16_k(
    const float* __restrict__ src, bf16_t* __restrict__ dst, int n) {
  int i = blockIdx.x * 256 + threadIdx.x;
  if (i < n) dst[i] = (bf16_t)src[i];
}

// ---------------------------------------------------------------------------
// Kernel 1b: f32 [R][C] -> bf16 transposed [C][R] (weights), LDS 32x32 tiles.
// ---------------------------------------------------------------------------
__global__ __launch_bounds__(256) void transpose_cvt_k(
    const float* __restrict__ src, bf16_t* __restrict__ dst, int R, int C) {
  __shared__ float tile[32][33];
  int c0 = blockIdx.x * 32, r0 = blockIdx.y * 32;
  int tx = threadIdx.x & 31, ty = threadIdx.x >> 5;
#pragma unroll
  for (int i = 0; i < 32; i += 8)
    tile[ty + i][tx] = src[(size_t)(r0 + ty + i) * C + c0 + tx];
  __syncthreads();
#pragma unroll
  for (int i = 0; i < 32; i += 8)
    dst[(size_t)(c0 + ty + i) * R + r0 + tx] = (bf16_t)tile[tx][ty + i];
}

// ---------------------------------------------------------------------------
// Kernel 2: bf16 GEMM  C[M,N] = A[M,K] @ Bt[N,K]^T (+bias), f32 accumulate.
// Block 128 thr = 4 waves; WG tile 128m x 128n; wave tile 32m x 128n.
// Bt staged in 64-wide K chunks, double-buffered; TDM or async-to-LDS.
// ---------------------------------------------------------------------------
template <typename OutT>
__global__ __launch_bounds__(128) void gemm_bf16_k(
    const bf16_t* __restrict__ A,   // [M][K]
    const bf16_t* __restrict__ Bt,  // [N][K]
    const float* __restrict__ bias, OutT* __restrict__ C,
    int M, int N, int K) {
  __shared__ bf16_t Bs[2][128][64];   // [buf][n][k], 2 x 16 KB

  const int wave = threadIdx.x >> 5;
  const int lane = threadIdx.x & 31;
  const int h2 = lane >> 4;
  const int ln = lane & 15;
  const int m0 = blockIdx.y * 128 + wave * 32;
  const int n0 = blockIdx.x * 128;

  v8f acc[2][8] = {};

  auto stage = [&](int buf, int k0) {
#if USE_TDM
    if (wave == 0)
      tdm_load_2d((unsigned)(uintptr_t)&Bs[buf][0][0],
                  Bt + (size_t)n0 * K + k0, /*x=*/64, /*y=*/128, /*stride=*/K);
#else
#pragma unroll
    for (int i = 0; i < 8; ++i) {
      int e = (threadIdx.x + i * 128) * 8;
      int r = e >> 6, c = e & 63;
      async_ld_b128(&Bs[buf][r][c], &Bt[(size_t)(n0 + r) * K + k0 + c]);
    }
#endif
  };
  auto stage_wait = [&]() {
#if USE_TDM
    if (wave == 0) wait_tensor0();
#else
    wait_async0();
#endif
    __syncthreads();
  };

  stage(0, 0);
  stage_wait();

  for (int k0 = 0; k0 < K; k0 += 64) {
    const int buf = (k0 >> 6) & 1;
    if (k0 + 64 < K) stage(buf ^ 1, k0 + 64);   // overlap DMA with compute

#pragma unroll
    for (int kc = 0; kc < 2; ++kc) {
      v16bf af[2];
#pragma unroll
      for (int mi = 0; mi < 2; ++mi) {
        const bf16_t* ap =
            &A[(size_t)(m0 + mi * 16 + ln) * K + k0 + kc * 32 + 8 * h2];
        ld_frag(af[mi], ap, ap + 16);
        if (kc == 0 && k0 + 128 < K)
          __builtin_prefetch(&A[(size_t)(m0 + mi * 16 + ln) * K + k0 + 128], 0, 0);
      }
#pragma unroll
      for (int nt = 0; nt < 8; ++nt) {
        v16bf bfr;
        const bf16_t* bp = &Bs[buf][nt * 16 + ln][kc * 32 + 8 * h2];
        ld_frag(bfr, bp, bp + 16);
#pragma unroll
        for (int mi = 0; mi < 2; ++mi)
          acc[mi][nt] = wmma_bf16(af[mi], bfr, acc[mi][nt]);
      }
    }
    stage_wait();
  }

  if (sizeof(OutT) == 2) {
    bf16_t* Cs = (bf16_t*)&Bs[0][0][0];   // reuse 32 KB for the 128x128 tile
#pragma unroll
    for (int mi = 0; mi < 2; ++mi)
#pragma unroll
      for (int nt = 0; nt < 8; ++nt) {
        int n = nt * 16 + ln;
        float bv = bias ? bias[n0 + n] : 0.0f;
#pragma unroll
        for (int r = 0; r < 8; ++r) {
          int m = wave * 32 + mi * 16 + r + 8 * h2;
          Cs[m * 128 + n] = (bf16_t)((float)acc[mi][nt][r] + bv);
        }
      }
    __syncthreads();
    bf16_t* Cg = (bf16_t*)C;
#pragma unroll
    for (int i = 0; i < 16; ++i) {
      int e = (threadIdx.x + i * 128) * 8;
      int r = e >> 7, c = e & 127;
      *(uint4*)&Cg[(size_t)(blockIdx.y * 128 + r) * N + n0 + c] =
          *(uint4*)&Cs[r * 128 + c];
    }
  } else {
#pragma unroll
    for (int mi = 0; mi < 2; ++mi)
#pragma unroll
      for (int nt = 0; nt < 8; ++nt) {
        int n = n0 + nt * 16 + ln;
        float bv = bias ? bias[n] : 0.0f;
#pragma unroll
        for (int r = 0; r < 8; ++r) {
          int m = m0 + mi * 16 + r + 8 * h2;
          C[(size_t)m * N + n] = (OutT)((float)acc[mi][nt][r] + bv);
        }
      }
  }
}

// ---------------------------------------------------------------------------
// Kernel 3a: RoPE + head-major reorder for Q/K (Q pre-scaled for exp2 softmax).
// ---------------------------------------------------------------------------
__global__ __launch_bounds__(128) void rope_reorder_k(
    const bf16_t* __restrict__ X, const float* __restrict__ cosb,
    const float* __restrict__ sinb, bf16_t* __restrict__ Y,
    int nheads, float scale) {
  using namespace cfg;
  int idx = blockIdx.x;
  int t = idx / nheads;
  int hh = idx % nheads;
  int b = t / S, s = t % S;
  int d = threadIdx.x;

  size_t base = (size_t)t * nheads * D + (size_t)hh * D;
  float x  = (float)X[base + d];
  float c  = cosb[(size_t)t * D + d];
  float sn = sinb[(size_t)t * D + d];
  int dp = (d < 64) ? d + 64 : d - 64;
  float xr = (float)X[base + dp];
  float rot = (d < 64) ? -xr : xr;
  Y[(((size_t)b * nheads + hh) * S + s) * D + d] =
      (bf16_t)((x * c + rot * sn) * scale);
}

// ---------------------------------------------------------------------------
// Kernel 3b: V reorder + transpose: [TOK, NKV*D] -> [B, NKV, D, S].
// ---------------------------------------------------------------------------
__global__ __launch_bounds__(256) void v_transpose_k(
    const bf16_t* __restrict__ X, bf16_t* __restrict__ Y) {
  using namespace cfg;
  __shared__ bf16_t t[64][136];
  int bh = blockIdx.y;
  int b = bh / NKV, kv = bh % NKV;
  int s0 = blockIdx.x * 64;
  for (int i = threadIdx.x; i < 64 * 16; i += 256) {
    int r = i >> 4, c = (i & 15) * 8;
    *(uint4*)&t[r][c] =
        *(const uint4*)&X[(size_t)(b * S + s0 + r) * (NKV * D) + kv * D + c];
  }
  __syncthreads();
  for (int i = threadIdx.x; i < 128 * 8; i += 256) {
    int d = i >> 3, c = (i & 7) * 8;
    uint4 val;
    bf16_t* vv = (bf16_t*)&val;
#pragma unroll
    for (int j = 0; j < 8; ++j) vv[j] = t[c + j][d];
    *(uint4*)&Y[(((size_t)(b * NKV + kv)) * D + d) * S + s0 + c] = val;
  }
}

// ---------------------------------------------------------------------------
// Kernel 4: causal flash attention (bf16 WMMA, exp2-domain online softmax).
// Grid (S/64, B*NH), block 128 = 4 waves; wave owns 16 q rows.
// K [kv][d] and V [d][kv] tiles double-buffered; TDM or async-to-LDS.
// ---------------------------------------------------------------------------
__global__ __launch_bounds__(128) void flash_attn_k(
    const bf16_t* __restrict__ Qh,  // [B,NH,S,D], pre-scaled by 1/sqrt(D)*log2e
    const bf16_t* __restrict__ Kh,  // [B,NKV,S,D]
    const bf16_t* __restrict__ Vt,  // [B,NKV,D,S]
    bf16_t* __restrict__ AO) {      // [TOK, NH*D]
  using namespace cfg;
  __shared__ bf16_t Ks[2][64][128];   // 2 x 16 KB
  __shared__ bf16_t Vs[2][128][64];   // 2 x 16 KB
  __shared__ bf16_t Ps[4][16][64];    // 8 KB per-wave P staging

  const int bh = blockIdx.y;
  const int b  = bh / NH;
  const int hh = bh % NH;
  const int kvh = hh / GROUPS;
  const int q0 = blockIdx.x * 64;
  const int wave = threadIdx.x >> 5;
  const int lane = threadIdx.x & 31;
  const int h2 = lane >> 4;
  const int ln = lane & 15;

  const bf16_t* Qb  = Qh + ((size_t)(b * NH  + hh ) * S) * D;
  const bf16_t* Kb  = Kh + ((size_t)(b * NKV + kvh) * S) * D;
  const bf16_t* Vtb = Vt + ((size_t)(b * NKV + kvh) * D) * S;

  v16bf qf[4];
  const int qrowA = q0 + 16 * wave + ln;
#pragma unroll
  for (int kc = 0; kc < 4; ++kc) {
    const bf16_t* qp = &Qb[(size_t)qrowA * D + kc * 32 + 8 * h2];
    ld_frag(qf[kc], qp, qp + 16);
  }

  v8f oacc[8] = {};
  float mrow[8], lrow[8];
#pragma unroll
  for (int r = 0; r < 8; ++r) { mrow[r] = NEG; lrow[r] = 0.0f; }

  auto stageKV = [&](int buf, int kv0) {
#if USE_TDM
    if (wave == 0) {
      tdm_load_2d((unsigned)(uintptr_t)&Ks[buf][0][0],
                  Kb + (size_t)kv0 * D, /*x=*/128, /*y=*/64, /*stride=*/D);
      tdm_load_2d((unsigned)(uintptr_t)&Vs[buf][0][0],
                  Vtb + kv0, /*x=*/64, /*y=*/128, /*stride=*/S);
    }
#else
#pragma unroll
    for (int i = 0; i < 8; ++i) {
      int e = (threadIdx.x + i * 128) * 8;
      { int r = e >> 7, c = e & 127;
        async_ld_b128(&Ks[buf][r][c], &Kb[(size_t)(kv0 + r) * D + c]); }
      { int d = e >> 6, c = e & 63;
        async_ld_b128(&Vs[buf][d][c], &Vtb[(size_t)d * S + kv0 + c]); }
    }
#endif
  };
  auto stage_wait = [&]() {
#if USE_TDM
    if (wave == 0) wait_tensor0();
#else
    wait_async0();
#endif
    __syncthreads();
  };

  const int kv_end  = q0 + 64;
  const int my_q_hi = q0 + 16 * wave + 15;

  stageKV(0, 0);
  stage_wait();

  for (int kv0 = 0; kv0 < kv_end; kv0 += 64) {
    const int buf = (kv0 >> 6) & 1;
    if (kv0 + 64 < kv_end) stageKV(buf ^ 1, kv0 + 64);

    if (kv0 <= my_q_hi) {
      // ---- S = Q @ K^T (exp2 domain) -------------------------------------
      v8f sc[4] = {};
#pragma unroll
      for (int kc = 0; kc < 4; ++kc)
#pragma unroll
        for (int nt = 0; nt < 4; ++nt) {
          v16bf bfr;
          const bf16_t* kp = &Ks[buf][nt * 16 + ln][kc * 32 + 8 * h2];
          ld_frag(bfr, kp, kp + 16);
          sc[nt] = wmma_bf16(qf[kc], bfr, sc[nt]);
        }
      // ---- causal mask: diagonal tiles only ------------------------------
      if (kv0 + 63 > q0 + 16 * wave) {
#pragma unroll
        for (int nt = 0; nt < 4; ++nt)
#pragma unroll
          for (int r = 0; r < 8; ++r) {
            int kvc = kv0 + nt * 16 + ln;
            int qm  = q0 + 16 * wave + r + 8 * h2;
            if (kvc > qm) sc[nt][r] = NEG;
          }
      }
      // ---- row max -------------------------------------------------------
      float tmax[8];
#pragma unroll
      for (int r = 0; r < 8; ++r) tmax[r] = NEG;
#pragma unroll
      for (int nt = 0; nt < 4; ++nt)
#pragma unroll
        for (int r = 0; r < 8; ++r) tmax[r] = fmaxf(tmax[r], (float)sc[nt][r]);
#pragma unroll
      for (int r = 0; r < 8; ++r)
#pragma unroll
        for (int off = 1; off < 16; off <<= 1)
          tmax[r] = fmaxf(tmax[r], __shfl_xor(tmax[r], off, 32));

      float alpha[8];
#pragma unroll
      for (int r = 0; r < 8; ++r) {
        float mnew = fmaxf(mrow[r], tmax[r]);
        alpha[r] = fast_exp2(mrow[r] - mnew);
        mrow[r] = mnew;
      }
      // ---- P = exp2(S - m), row sums, stage P ----------------------------
      float lsum[8];
#pragma unroll
      for (int r = 0; r < 8; ++r) lsum[r] = 0.0f;
#pragma unroll
      for (int nt = 0; nt < 4; ++nt)
#pragma unroll
        for (int r = 0; r < 8; ++r) {
          float p = fast_exp2((float)sc[nt][r] - mrow[r]);
          lsum[r] += p;
          Ps[wave][r + 8 * h2][nt * 16 + ln] = (bf16_t)p;
        }
#pragma unroll
      for (int r = 0; r < 8; ++r) {
#pragma unroll
        for (int off = 1; off < 16; off <<= 1)
          lsum[r] += __shfl_xor(lsum[r], off, 32);
        lrow[r] = lrow[r] * alpha[r] + lsum[r];
      }
#pragma unroll
      for (int t = 0; t < 8; ++t)
#pragma unroll
        for (int r = 0; r < 8; ++r) oacc[t][r] *= alpha[r];

      // ---- O += P @ V ----------------------------------------------------
      v16bf pf[2];
#pragma unroll
      for (int kc = 0; kc < 2; ++kc) {
        const bf16_t* pp = &Ps[wave][ln][kc * 32 + 8 * h2];
        ld_frag(pf[kc], pp, pp + 16);
      }
#pragma unroll
      for (int kc = 0; kc < 2; ++kc)
#pragma unroll
        for (int dt = 0; dt < 8; ++dt) {
          v16bf vfr;
          const bf16_t* vp = &Vs[buf][dt * 16 + ln][kc * 32 + 8 * h2];
          ld_frag(vfr, vp, vp + 16);
          oacc[dt] = wmma_bf16(pf[kc], vfr, oacc[dt]);
        }
    }
    stage_wait();
  }

  // ---- finalize: O *= 1/l; stage in LDS (reuse Ks), store b128 -----------
  float linv[8];
#pragma unroll
  for (int r = 0; r < 8; ++r) linv[r] = fast_rcp(lrow[r]);
  bf16_t* Osh = &Ks[0][0][0];   // 64 x 128 bf16 = 16 KB
#pragma unroll
  for (int dt = 0; dt < 8; ++dt)
#pragma unroll
    for (int r = 0; r < 8; ++r) {
      float o = (float)oacc[dt][r] * linv[r];
      Osh[(wave * 16 + r + 8 * h2) * 128 + dt * 16 + ln] = (bf16_t)o;
    }
  __syncthreads();
#pragma unroll
  for (int i = 0; i < 8; ++i) {
    int e = (threadIdx.x + i * 128) * 8;
    int r = e >> 7, c = e & 127;
    *(uint4*)&AO[((size_t)(b * S + q0 + r)) * (NH * D) + hh * D + c] =
        *(uint4*)&Osh[r * 128 + c];
  }
}

// ---------------------------------------------------------------------------
// Host launcher
// ---------------------------------------------------------------------------
extern "C" void kernel_launch(void* const* d_in, const int* in_sizes, int n_in,
                              void* d_out, int out_size, void* d_ws, size_t ws_size,
                              hipStream_t stream) {
  using namespace cfg;
  const float* hidden = (const float*)d_in[0];
  const float* cosb   = (const float*)d_in[1];
  const float* sinb   = (const float*)d_in[2];
  const float* Wq = (const float*)d_in[3];
  const float* bq = (const float*)d_in[4];
  const float* Wk = (const float*)d_in[5];
  const float* bk = (const float*)d_in[6];
  const float* Wv = (const float*)d_in[7];
  const float* bv = (const float*)d_in[8];
  const float* Wo = (const float*)d_in[9];
  float* out = (float*)d_out;

  char* wsp = (char*)d_ws;
  size_t off = 0;
  auto slab = [&](size_t bytes) {
    char* p = wsp + off;
    off += bytes;
    off = (off + 255) & ~(size_t)255;
    return p;
  };
  bf16_t* hb  = (bf16_t*)slab((size_t)TOK * H * 2);
  bf16_t* wqt = (bf16_t*)slab((size_t)NH  * D * H * 2);
  bf16_t* wkt = (bf16_t*)slab((size_t)NKV * D * H * 2);
  bf16_t* wvt = (bf16_t*)slab((size_t)NKV * D * H * 2);
  bf16_t* wot = (bf16_t*)slab((size_t)H * NH * D * 2);
  bf16_t* qt  = (bf16_t*)slab((size_t)TOK * NH  * D * 2);
  bf16_t* kt  = (bf16_t*)slab((size_t)TOK * NKV * D * 2);
  bf16_t* vt  = (bf16_t*)slab((size_t)TOK * NKV * D * 2);
  bf16_t* qh  = (bf16_t*)slab((size_t)TOK * NH  * D * 2);
  bf16_t* kh  = (bf16_t*)slab((size_t)TOK * NKV * D * 2);
  bf16_t* vh  = (bf16_t*)slab((size_t)TOK * NKV * D * 2);
  bf16_t* ao  = (bf16_t*)slab((size_t)TOK * NH  * D * 2);
  (void)ws_size; (void)in_sizes; (void)n_in; (void)out_size;

  // 1) conversions (+weight transposes)
  cvt_bf16_k<<<(TOK * H + 255) / 256, 256, 0, stream>>>(hidden, hb, TOK * H);
  transpose_cvt_k<<<dim3((NH  * D) / 32, H / 32), 256, 0, stream>>>(Wq, wqt, H, NH  * D);
  transpose_cvt_k<<<dim3((NKV * D) / 32, H / 32), 256, 0, stream>>>(Wk, wkt, H, NKV * D);
  transpose_cvt_k<<<dim3((NKV * D) / 32, H / 32), 256, 0, stream>>>(Wv, wvt, H, NKV * D);
  transpose_cvt_k<<<dim3(H / 32, (NH * D) / 32), 256, 0, stream>>>(Wo, wot, NH * D, H);

  // 2) QKV projections
  dim3 blk(128);
  gemm_bf16_k<bf16_t><<<dim3((NH  * D) / 128, TOK / 128), blk, 0, stream>>>(
      hb, wqt, bq, qt, TOK, NH * D, H);
  gemm_bf16_k<bf16_t><<<dim3((NKV * D) / 128, TOK / 128), blk, 0, stream>>>(
      hb, wkt, bk, kt, TOK, NKV * D, H);
  gemm_bf16_k<bf16_t><<<dim3((NKV * D) / 128, TOK / 128), blk, 0, stream>>>(
      hb, wvt, bv, vt, TOK, NKV * D, H);

  // 3) RoPE (+Q pre-scale) and reorders
  rope_reorder_k<<<TOK * NH,  D, 0, stream>>>(qt, cosb, sinb, qh, NH,  QSCALE);
  rope_reorder_k<<<TOK * NKV, D, 0, stream>>>(kt, cosb, sinb, kh, NKV, 1.0f);
  v_transpose_k<<<dim3(S / 64, B * NKV), 256, 0, stream>>>(vt, vh);

  // 4) causal flash attention
  flash_attn_k<<<dim3(S / 64, B * NH), blk, 0, stream>>>(qh, kh, vh, ao);

  // 5) output projection (f32 out)
  gemm_bf16_k<float><<<dim3(H / 128, TOK / 128), blk, 0, stream>>>(
      ao, wot, nullptr, out, TOK, H, NH * D);
}
